// FeatureAdapter_55791625175135
// MI455X (gfx1250) — matrix operand order repaired and checked by reading, not compiled, not run on hardware
//
#include <hip/hip_runtime.h>
#include <hip/hip_bf16.h>
#include <stdint.h>

typedef __attribute__((ext_vector_type(16))) __bf16 v16bf;
typedef __attribute__((ext_vector_type(8)))  float  v8f;

#define N_ROWS 16384
#define H_HEADS 8
#define D_DIM 1024
#define BM 128
#define BN 128
// A panel in LDS: 32 k-tiles x (BM/16 = 8) row-tiles x 512 bf16 = 256KB (k-major!)
#define PANEL_USHORTS (32 * (BM / 16) * 512)
// + Ss row-major (4KB) + Sst transposed (4KB)
#define SMEM_BYTES (PANEL_USHORTS * 2 + BM * 8 * 4 + BM * 8 * 4)

union ABf { uint4 q[2]; v16bf v; };

static __device__ __forceinline__ unsigned short f2bf(float f) {
    unsigned u = __float_as_uint(f);
    u += 0x7FFFu + ((u >> 16) & 1u);   // round-to-nearest-even
    return (unsigned short)(u >> 16);
}

// ---------------- x f32 -> bf16, pre-swizzled into WMMA A-fragment tile order ----------------
// Global layout: [rbGroup = n/128][kb][rbl = (n/16)%8][512], so each 128-row panel is a
// contiguous 256KB block in k-major order (matches the LDS panel layout exactly).
// Value slot = lane*16 + ev, lane = half*16 + m,
// ev per ISA A(16x32) layout: K = ev<8 ? half*8+ev : 16+half*8+(ev-8).
__global__ void k_swz_x(const float* __restrict__ x, unsigned short* __restrict__ xswz) {
    const int n  = blockIdx.x;
    const int d0 = threadIdx.x * 4;
    float4 v = *(const float4*)(x + (size_t)n * D_DIM + d0);
    const int m     = n & 15;
    const int rbl   = (n >> 4) & 7;
    const int group = n >> 7;
    const int kb    = d0 >> 5;
    const int dl    = d0 & 31;
    const int half  = (dl >> 3) & 1;
    const int ev0   = (dl & 7) + ((dl & 16) ? 8 : 0);
    const int lane  = half * 16 + m;
    const size_t o = (((size_t)group * 32 + kb) * 8 + rbl) * 512 + lane * 16 + ev0;
    uint2 p;
    p.x = (unsigned)f2bf(v.x) | ((unsigned)f2bf(v.y) << 16);
    p.y = (unsigned)f2bf(v.z) | ((unsigned)f2bf(v.w) << 16);
    *(uint2*)(xswz + o) = p;
}

// ---------------- W[h,e,d] f32 -> bf16, pre-swizzled into WMMA B-fragment tile order ----------------
// B tile (h, kb, eb): 32x16, lane = half*16 + col, elem K = e<8 ? half*8+e : 16+half*8+(e-8).
__global__ void k_prep_w(const float* __restrict__ W, unsigned short* __restrict__ Wt) {
    __shared__ float tile[32][33];
    const int h  = blockIdx.z;
    const int d0 = blockIdx.x * 32, e0 = blockIdx.y * 32;
    const int tx = threadIdx.x & 31, ty = threadIdx.x >> 5;  // ty: 0..7
    const float* Wh = W + (size_t)h * D_DIM * D_DIM;
#pragma unroll
    for (int j = 0; j < 32; j += 8)
        tile[ty + j][tx] = Wh[(size_t)(e0 + ty + j) * D_DIM + d0 + tx];
    __syncthreads();
    // writer: eL = local col 0..31, kg*4 = local k
    const int eL  = threadIdx.x & 31;
    const int kg  = threadIdx.x >> 5;              // 0..7
    const int kl0 = kg * 4;
    const int lane  = ((kg >> 1) & 1) * 16 + (eL & 15);  // half = (kl0>>3)&1
    const int elem0 = (kg & 1) * 4 + (kg >> 2) * 8;      // (kl0&7) + (kl0&16 ? 8 : 0)
    const size_t tileId = ((size_t)(h * 32 + (d0 >> 5))) * 64 + (e0 >> 4) + (eL >> 4);
    uint2 p;
    p.x = (unsigned)f2bf(tile[eL][kl0 + 0]) | ((unsigned)f2bf(tile[eL][kl0 + 1]) << 16);
    p.y = (unsigned)f2bf(tile[eL][kl0 + 2]) | ((unsigned)f2bf(tile[eL][kl0 + 3]) << 16);
    *(uint2*)(Wt + tileId * 512 + lane * 16 + elem0) = p;
}

// ---------------- cosine-sim + softmax: one wave32 per row ----------------
__global__ void k_softmax(const float* __restrict__ x, const float* __restrict__ P,
                          float* __restrict__ s_out) {
    const int wave = threadIdx.x >> 5;
    const int lane = threadIdx.x & 31;
    const int n = blockIdx.x * 8 + wave;
    const float* xr = x + (size_t)n * D_DIM;

    float dot[8], pp[8], xx = 0.f;
#pragma unroll
    for (int h = 0; h < 8; h++) { dot[h] = 0.f; pp[h] = 0.f; }
    for (int d = lane; d < D_DIM; d += 32) {
        float xv = xr[d];
        xx += xv * xv;
#pragma unroll
        for (int h = 0; h < 8; h++) {
            float pv = P[h * D_DIM + d];
            dot[h] += xv * pv;
            pp[h]  += pv * pv;
        }
    }
#pragma unroll
    for (int off = 16; off > 0; off >>= 1) {
        xx += __shfl_xor(xx, off, 32);
#pragma unroll
        for (int h = 0; h < 8; h++) {
            dot[h] += __shfl_xor(dot[h], off, 32);
            pp[h]  += __shfl_xor(pp[h],  off, 32);
        }
    }
    const float xn = sqrtf(xx);
    float sh[8], m = -1e30f;
#pragma unroll
    for (int h = 0; h < 8; h++) {
        sh[h] = dot[h] / fmaxf(xn * sqrtf(pp[h]), 1e-8f);
        m = fmaxf(m, sh[h]);
    }
    float e[8], sum = 0.f;
#pragma unroll
    for (int h = 0; h < 8; h++) { e[h] = __expf((sh[h] - m) * 10.0f); sum += e[h]; }
    const float inv = 1.0f / sum;
    if (lane == 0) {
#pragma unroll
        for (int h = 0; h < 8; h++) s_out[(size_t)n * 8 + h] = e[h] * inv;
    }
}

// ---------------- main GEMM: barrier-free K-loop, 64x32 wave tile, k-major LDS panel ----------------
__global__ __launch_bounds__(256, 1) void k_gemm(
    const unsigned short* __restrict__ xswz,  // [N/128][32][8][512] bf16, A-frag order
    const unsigned short* __restrict__ Wt,    // [H*32*64][512] bf16, B-frag order
    const float* __restrict__ sw,             // [N, 8]
    const float* __restrict__ x,              // [N, D] f32 (residual)
    const float* __restrict__ bias,           // [H, D] f32
    float* __restrict__ out)                  // [N, D] f32
{
    extern __shared__ char smem[];
    unsigned short* Ap = (unsigned short*)smem;                    // 256KB A panel [kk][rbl][512]
    float* Ss  = (float*)(smem + PANEL_USHORTS * 2);               // [row][h] 4KB
    float* Sst = (float*)(smem + PANEL_USHORTS * 2 + BM * 8 * 4);  // [h][row] 4KB

    const int tid = threadIdx.x;
    const int n0 = blockIdx.y * BM;
    const int e0 = blockIdx.x * BN;

    // stage whole A panel (contiguous, already fragment-and-k-major ordered) into LDS once
    {
        const uint4* src = (const uint4*)(xswz + (size_t)(n0 >> 7) * PANEL_USHORTS);
        uint4* dst = (uint4*)Ap;
#pragma unroll 4
        for (int i = tid; i < PANEL_USHORTS / 8; i += 256) dst[i] = src[i];
    }
    // stage softmax weights: row-major (for epilogue) and transposed (for head fold)
    *(float4*)(Ss + tid * 4) = *(const float4*)(sw + (size_t)n0 * 8 + tid * 4);
#pragma unroll
    for (int idx = 0; idx < 4; idx++) {
        const int v = tid + idx * 256;
        const int hh = v >> 7, row = v & 127;
        Sst[hh * BM + row] = sw[(size_t)(n0 + row) * 8 + hh];
    }
    __syncthreads();

    const int wave = tid >> 5;
    const int lane = tid & 31;
    const int wm = (wave & 1) * 64;        // wave row offset: 0/64 (4 row-tiles each)
    const int wn = (wave >> 1) * 32;       // wave col offset: 0/32/64/96 (2 col-tiles)
    const int kh = lane >> 4;
    const int l15 = lane & 15;
    const int rb0 = wm >> 4;               // local A row-tile base for this wave (0 or 4)
    const int ebW = ((e0 + wn) >> 4);      // global B col-tile base for this wave

    const v8f vzero = {0.f, 0.f, 0.f, 0.f, 0.f, 0.f, 0.f, 0.f};
    v8f acc[4][2];
#pragma unroll
    for (int i = 0; i < 4; i++)
#pragma unroll
        for (int j = 0; j < 2; j++) acc[i][j] = vzero;

    for (int h = 0; h < H_HEADS; ++h) {
        v8f pacc[4][2];
#pragma unroll
        for (int i = 0; i < 4; i++)
#pragma unroll
            for (int j = 0; j < 2; j++) pacc[i][j] = vzero;

#pragma unroll 2
        for (int kk = 0; kk < 32; ++kk) {
            // all 4 A frags for this kk share one LDS base; per-frag offsets fit the
            // 16-bit DS immediate (i*1024 + {0,16} bytes)
            const unsigned short* abase = Ap + (kk * 8 + rb0) * 512 + lane * 16;
            v16bf afrag[4], bfrag[2];
#pragma unroll
            for (int j = 0; j < 2; j++) {
                const unsigned short* bp =
                    Wt + ((size_t)((h * 32 + kk) * 64 + ebW + j)) * 512 + lane * 16;
                ABf u;
                u.q[0] = *(const uint4*)bp;
                u.q[1] = *(const uint4*)(bp + 8);
                bfrag[j] = u.v;
            }
#pragma unroll
            for (int i = 0; i < 4; i++) {
                ABf u;
                u.q[0] = *(const uint4*)(abase + i * 512);
                u.q[1] = *(const uint4*)(abase + i * 512 + 8);
                afrag[i] = u.v;
            }
#pragma unroll
            for (int i = 0; i < 4; i++)
#pragma unroll
                for (int j = 0; j < 2; j++)
                    pacc[i][j] = __builtin_amdgcn_wmma_f32_16x16x32_bf16(
                        false, afrag[i], false, bfrag[j], (short)0, pacc[i][j], false, false);
        }
        // fold this head into the running accumulator with its softmax weight
#pragma unroll
        for (int i = 0; i < 4; i++) {
            float sarr[8];
            *(float4*)&sarr[0] = *(const float4*)(Sst + h * BM + wm + i * 16 + kh * 8);
            *(float4*)&sarr[4] = *(const float4*)(Sst + h * BM + wm + i * 16 + kh * 8 + 4);
#pragma unroll
            for (int r = 0; r < 8; r++) {
                acc[i][0][r] += sarr[r] * pacc[i][0][r];
                acc[i][1][r] += sarr[r] * pacc[i][1][r];
            }
        }
    }

    // epilogue: residual + softmax-mixed bias
#pragma unroll
    for (int j = 0; j < 2; j++) {
        const int e = e0 + wn + j * 16 + l15;
        float bb[8];
#pragma unroll
        for (int h = 0; h < 8; h++) bb[h] = bias[h * D_DIM + e];
#pragma unroll
        for (int i = 0; i < 4; i++) {
#pragma unroll
            for (int r = 0; r < 8; r++) {
                const int row = wm + i * 16 + kh * 8 + r;
                float bm = 0.f;
#pragma unroll
                for (int h = 0; h < 8; h++) bm += Ss[row * 8 + h] * bb[h];
                const size_t off = (size_t)(n0 + row) * D_DIM + e;
                out[off] = x[off] + acc[i][j][r] + bm;
            }
        }
    }
}

extern "C" void kernel_launch(void* const* d_in, const int* in_sizes, int n_in,
                              void* d_out, int out_size, void* d_ws, size_t ws_size,
                              hipStream_t stream) {
    const float* x = (const float*)d_in[0];
    const float* P = (const float*)d_in[1];
    const float* W = (const float*)d_in[2];
    const float* b = (const float*)d_in[3];
    float* out = (float*)d_out;

    char* ws = (char*)d_ws;
    unsigned short* xswz = (unsigned short*)ws;                        // 32 MB
    unsigned short* Wt   = (unsigned short*)(ws + ((size_t)32 << 20)); // 16 MB
    float*          sw   = (float*)(ws + ((size_t)48 << 20));          // 512 KB

    (void)hipFuncSetAttribute((const void*)k_gemm,
                              hipFuncAttributeMaxDynamicSharedMemorySize, SMEM_BYTES);

    k_swz_x  <<<dim3(N_ROWS),      256, 0, stream>>>(x, xswz);
    k_prep_w <<<dim3(32, 32, 8),   256, 0, stream>>>(W, Wt);
    k_softmax<<<dim3(N_ROWS / 8),  256, 0, stream>>>(x, P, sw);
    k_gemm   <<<dim3(D_DIM / BN, N_ROWS / BM), 256, SMEM_BYTES, stream>>>(
        xswz, Wt, sw, x, b, out);
}
